// SpatialGNNDensity_4363686773364
// MI455X (gfx1250) — matrix operand order repaired and checked by reading, not compiled, not run on hardware
//
#include <hip/hip_runtime.h>

// ---------------------------------------------------------------------------
// SpatialGNNDensity on MI455X (gfx1250): bf16 WMMA GEMMs + grid-stencil GCN.
// LEVELS = (256,256),(128,128),(64,64) ; DIM=192 ; HID=256 ; NTOT=86016
// ---------------------------------------------------------------------------

typedef __attribute__((ext_vector_type(16))) __bf16 v16bf;
typedef __attribute__((ext_vector_type(8)))  __bf16 v8bf;
typedef __attribute__((ext_vector_type(4)))  __bf16 v4bf;
typedef __attribute__((ext_vector_type(8)))  float  v8f;

#define NTOT 86016
#define HID  256

// gfx1250 async global->LDS path (ASYNCcnt-tracked, no VGPR bounce)
#if defined(__has_builtin)
#if __has_builtin(__builtin_amdgcn_global_load_async_to_lds_b128)
#define USE_ASYNC_LDS 1
#endif
#endif

#ifdef USE_ASYNC_LDS
typedef int vi4 __attribute__((vector_size(16)));
typedef __attribute__((address_space(1))) vi4 g_vi4;   // global int4
typedef __attribute__((address_space(3))) vi4 l_vi4;   // LDS int4

template <int OFF>
__device__ __forceinline__ void async_cp16(const __bf16* g, __bf16* l) {
  // GLOBAL_LOAD_ASYNC_TO_LDS_B128: IOFFSET applies to both global and LDS addr
  __builtin_amdgcn_global_load_async_to_lds_b128((g_vi4*)g, (l_vi4*)l, OFF, 0);
}

__device__ __forceinline__ void wait_asynccnt0() {
#if __has_builtin(__builtin_amdgcn_s_wait_asynccnt)
  __builtin_amdgcn_s_wait_asynccnt(0);
#else
  asm volatile("s_wait_asynccnt 0" ::: "memory");
#endif
}
#endif

// ---------------------------------------------------------------------------
// Pack f32 weight [K,256] into bf16 WMMA B-fragment order.
// Fragment (nt, kt) covers cols nt*16..+15, k kt*32..+31.
// B layout used here (contiguous k, same as A): lane l<16 -> col=nt*16+l, k = kt*32 + e (e=0..15);
// lane l>=16 -> col=nt*16+(l-16), k = kt*32 + 16 + e. 16 contiguous bf16/lane.
// ---------------------------------------------------------------------------
__global__ void pack_weight(const float* __restrict__ W, __bf16* __restrict__ Wp, int K) {
  int tid = blockIdx.x * blockDim.x + threadIdx.x;      // one thread per (frag,lane)
  int K32 = K >> 5;
  int nfrag = K32 * (HID / 16);
  if (tid >= nfrag * 32) return;
  int frag = tid >> 5;
  int lane = tid & 31;
  int kt = frag % K32;
  int nt = frag / K32;
  int n  = nt * 16 + (lane & 15);
  int kb = kt * 32 + ((lane >> 4) << 4);
  v16bf o;
#pragma unroll
  for (int e = 0; e < 16; ++e)
    o[e] = (__bf16)W[(size_t)(kb + e) * HID + n];
  *(v16bf*)(Wp + (size_t)frag * 512 + lane * 16) = o;
}

// ---------------------------------------------------------------------------
// NCHW f32 features -> row-major bf16 node features via LDS tile transpose.
// Tile: 32 channels x 64 pixels. grid = (HW/64, DIM/32), block = 256.
// ---------------------------------------------------------------------------
__global__ __launch_bounds__(256) void transpose_feats(
    const float* __restrict__ F, __bf16* __restrict__ Xb, int HWn, int base) {
  __shared__ float tile[32][65];
  int p0 = blockIdx.x * 64;
  int c0 = blockIdx.y * 32;
  int t  = threadIdx.x;
  int r  = t >> 3;            // channel row 0..31
  int cs = (t & 7) * 8;       // pixel chunk
  const float* src = F + (size_t)(c0 + r) * HWn + p0 + cs;
#pragma unroll
  for (int j = 0; j < 8; ++j) tile[r][cs + j] = src[j];
  __syncthreads();
  int np = t >> 2;            // node 0..63
  int cc = (t & 3) * 8;       // channel chunk
  v8bf o;
#pragma unroll
  for (int j = 0; j < 8; ++j) o[j] = (__bf16)tile[cc + j][np];
  *(v8bf*)(Xb + (size_t)(base + p0 + np) * 192 + c0 + cc) = o;
}

// ---------------------------------------------------------------------------
// GEMM: Y[M,256] = A[M,K](bf16) @ W[K,256](packed bf16) (+bias)(+relu)
// Block tile 128x128, 8 waves in 4x2, each wave 32x64 = 2x4 WMMA 16x16 tiles.
// Double-buffered LDS A staging via GLOBAL_LOAD_ASYNC_TO_LDS_B128 (ASYNCcnt),
// 1 barrier / K-step; B fragments straight from packed global (L2-resident).
// All epilogue options are template parameters -> branch-free straight-line.
// ---------------------------------------------------------------------------
template <int K, bool BIAS, bool RELU, bool OUTF>
__global__ __launch_bounds__(256) void gemm_wmma(
    const __bf16* __restrict__ A, const __bf16* __restrict__ Wp,
    const float* __restrict__ bias, void* __restrict__ outp, int M) {
  constexpr int LDA = 40;                 // bf16 elems per LDS row (80B, padded)
  constexpr int K32 = K >> 5;
  __shared__ __bf16 sA[2][128 * LDA];

  const int tid  = threadIdx.x;
  const int lane = tid & 31;
  const int wave = tid >> 5;
  const int mi   = wave >> 1;             // 0..3 : 32-row slabs
  const int ni   = wave & 1;              // 0..1 : 64-col slabs
  const int rowBlock = blockIdx.x * 128;
  const int colBase  = blockIdx.y * 128 + ni * 64;

  v8f acc[2][4] = {};

  // staging: 2 threads per row, 16 bf16 (32B) each
  const int sRow  = tid >> 1;
  const int sHalf = (tid & 1) * 16;
  const __bf16* gaBase = A + (size_t)(rowBlock + sRow) * K + sHalf;

#ifdef USE_ASYNC_LDS
  {
    __bf16* lp = &sA[0][sRow * LDA + sHalf];
    async_cp16<0>(gaBase, lp);
    async_cp16<16>(gaBase, lp);
    wait_asynccnt0();
  }
  __syncthreads();
#else
  {
    uint4 r0 = *(const uint4*)gaBase;
    uint4 r1 = *(const uint4*)(gaBase + 8);
    *(uint4*)&sA[0][sRow * LDA + sHalf]     = r0;
    *(uint4*)&sA[0][sRow * LDA + sHalf + 8] = r1;
  }
  __syncthreads();
#endif

  const int koff  = (lane >> 4) * 16;     // lanes 16-31 take k 16..31
  const int laneM = lane & 15;

#ifndef USE_ASYNC_LDS
  uint4 r0, r1;
#endif

#pragma unroll
  for (int kt = 0; kt < K32; ++kt) {
    const int cur = kt & 1;
    // ---- fire next A tile toward the other LDS buffer (overlaps WMMA) ----
    if (kt + 1 < K32) {
#ifdef USE_ASYNC_LDS
      const __bf16* gp = gaBase + (kt + 1) * 32;
      __bf16* lp = &sA[cur ^ 1][sRow * LDA + sHalf];
      async_cp16<0>(gp, lp);
      async_cp16<16>(gp, lp);
#else
      const __bf16* ga = gaBase + (kt + 1) * 32;
      r0 = *(const uint4*)ga;
      r1 = *(const uint4*)(ga + 8);
#endif
    }
    // ---- B fragments from packed weights (global, hot in cache) ----
    v16bf bfrag[4];
#pragma unroll
    for (int j = 0; j < 4; ++j) {
      int nt = (colBase + j * 16) >> 4;
      const v16bf* bp = (const v16bf*)(Wp + (size_t)(nt * K32 + kt) * 512);
      bfrag[j] = bp[lane];
    }
    // ---- A fragments from LDS (contiguous k = 16h + e, same as packed B) ----
    v16bf afrag[2];
#pragma unroll
    for (int i = 0; i < 2; ++i) {
      const __bf16* sp = &sA[cur][(mi * 32 + i * 16 + laneM) * LDA];
      v8bf lo = *(const v8bf*)(sp + koff);
      v8bf hi = *(const v8bf*)(sp + koff + 8);
      afrag[i] = __builtin_shufflevector(lo, hi, 0, 1, 2, 3, 4, 5, 6, 7,
                                         8, 9, 10, 11, 12, 13, 14, 15);
    }
    // ---- 8x v_wmma_f32_16x16x32_bf16 ----
#pragma unroll
    for (int i = 0; i < 2; ++i)
#pragma unroll
      for (int j = 0; j < 4; ++j)
        acc[i][j] = __builtin_amdgcn_wmma_f32_16x16x32_bf16(
            false, afrag[i], false, bfrag[j], (short)0, acc[i][j], false, false);
    // ---- ensure next tile landed in LDS, then block-wide barrier ----
    if (kt + 1 < K32) {
#ifdef USE_ASYNC_LDS
      wait_asynccnt0();
#else
      *(uint4*)&sA[cur ^ 1][sRow * LDA + sHalf]     = r0;
      *(uint4*)&sA[cur ^ 1][sRow * LDA + sHalf + 8] = r1;
#endif
    }
    __syncthreads();
  }

  // ---- epilogue: branch-free bias / relu / store ----
  const int laneN = lane & 15;
  const int rAdd  = (lane >> 4) * 8;      // C/D layout: lanes 16-31 -> M+8
  float* __restrict__  Yf = (float*)outp;
  __bf16* __restrict__ Yb = (__bf16*)outp;
#pragma unroll
  for (int j = 0; j < 4; ++j) {
    const int col = colBase + j * 16 + laneN;
    const float bv = BIAS ? bias[col] : 0.0f;
#pragma unroll
    for (int i = 0; i < 2; ++i) {
      const int rbase = rowBlock + mi * 32 + i * 16 + rAdd;
#pragma unroll
      for (int v = 0; v < 8; ++v) {
        float r = acc[i][j][v] + bv;
        if (RELU) r = fmaxf(r, 0.0f);
        const size_t off = (size_t)(rbase + v) * HID + col;
        if (OUTF) Yf[off] = r;
        else      Yb[off] = (__bf16)r;
      }
    }
  }
  (void)M;
}

// ---------------------------------------------------------------------------
// GCN propagate as weighted 5-point stencil (grid graph + self loops).
// deg(node) = 1 + #in-bounds 4-neighbors ; norm(s,t) = rsqrt(deg_s)*rsqrt(deg_t)
// out = sum(norm * hw[nbr]) + dinv^2 * hw[self] + bias -> bf16 (next layer A).
// ---------------------------------------------------------------------------
__device__ __forceinline__ void stencil_add(
    float4& acc, float ds, const float* __restrict__ HWf,
    size_t nbrRow, int nx, int ny, int W, int H, int c) {
  float cn = 1.0f + (nx > 0) + (nx < W - 1) + (ny > 0) + (ny < H - 1);
  float wn = ds * rsqrtf(cn);
  const float4 b = *(const float4*)(HWf + nbrRow * HID + c);
  acc.x += wn * b.x; acc.y += wn * b.y; acc.z += wn * b.z; acc.w += wn * b.w;
}

__global__ __launch_bounds__(256) void gcn_aggregate(
    const float* __restrict__ HWf, const float* __restrict__ bg,
    __bf16* __restrict__ Hout, int H, int W, int base) {
  int tid = blockIdx.x * blockDim.x + threadIdx.x;
  int nodeL = tid >> 6;                   // 64 threads / node (4 ch each)
  int c = (tid & 63) * 4;
  if (nodeL >= H * W) return;
  int y = nodeL / W, x = nodeL % W;
  float cs = 1.0f + (x > 0) + (x < W - 1) + (y > 0) + (y < H - 1);
  float ds = rsqrtf(cs);
  size_t selfRow = (size_t)(base + nodeL);
  float4 a = *(const float4*)(HWf + selfRow * HID + c);
  float ws = ds * ds;
  float4 acc = {a.x * ws, a.y * ws, a.z * ws, a.w * ws};
  if (x > 0)     stencil_add(acc, ds, HWf, selfRow - 1, x - 1, y, W, H, c);
  if (x < W - 1) stencil_add(acc, ds, HWf, selfRow + 1, x + 1, y, W, H, c);
  if (y > 0)     stencil_add(acc, ds, HWf, selfRow - W, x, y - 1, W, H, c);
  if (y < H - 1) stencil_add(acc, ds, HWf, selfRow + W, x, y + 1, W, H, c);
  float4 bb = *(const float4*)(bg + c);
  v4bf o;
  o[0] = (__bf16)(acc.x + bb.x); o[1] = (__bf16)(acc.y + bb.y);
  o[2] = (__bf16)(acc.z + bb.z); o[3] = (__bf16)(acc.w + bb.w);
  *(v4bf*)(Hout + selfRow * HID + c) = o;
}

// ---------------------------------------------------------------------------
// Head layer 2: out[node] = dot(P[node,0:256], Wh2) + bh2 ; wave per node.
// ---------------------------------------------------------------------------
__global__ __launch_bounds__(256) void head_gemv(
    const __bf16* __restrict__ P, const float* __restrict__ w2,
    const float* __restrict__ b2, float* __restrict__ out, int N) {
  int wave = threadIdx.x >> 5, lane = threadIdx.x & 31;
  int node = blockIdx.x * 8 + wave;
  if (node >= N) return;
  const v8bf pv = *(const v8bf*)(P + (size_t)node * HID + lane * 8);
  float s = 0.0f;
#pragma unroll
  for (int j = 0; j < 8; ++j) s += (float)pv[j] * w2[lane * 8 + j];
#pragma unroll
  for (int off = 16; off > 0; off >>= 1) s += __shfl_xor(s, off, 32);
  if (lane == 0) out[node] = s + b2[0];
}

// ---------------------------------------------------------------------------
// Host-side orchestration (all launches on `stream`; scratch carved from d_ws)
// ---------------------------------------------------------------------------
extern "C" void kernel_launch(void* const* d_in, const int* in_sizes, int n_in,
                              void* d_out, int out_size, void* d_ws, size_t ws_size,
                              hipStream_t stream) {
  (void)in_sizes; (void)n_in; (void)out_size; (void)ws_size;
  static const int LH[3]   = {256, 128, 64};
  static const int LW[3]   = {256, 128, 64};
  static const int LSZ[3]  = {65536, 16384, 4096};
  static const int LOFF[3] = {0, 65536, 81920};

  const float* F[3] = {(const float*)d_in[0], (const float*)d_in[1], (const float*)d_in[2]};
  // d_in[3]=src, d_in[4]=dst : grid structure is known analytically -> unused.
  const float* eW1[3] = {(const float*)d_in[5],  (const float*)d_in[9],  (const float*)d_in[13]};
  const float* eB1[3] = {(const float*)d_in[6],  (const float*)d_in[10], (const float*)d_in[14]};
  const float* eW2[3] = {(const float*)d_in[7],  (const float*)d_in[11], (const float*)d_in[15]};
  const float* eB2[3] = {(const float*)d_in[8],  (const float*)d_in[12], (const float*)d_in[16]};
  const float* gW[3]  = {(const float*)d_in[17], (const float*)d_in[19], (const float*)d_in[21]};
  const float* gB[3]  = {(const float*)d_in[18], (const float*)d_in[20], (const float*)d_in[22]};
  const float* Wh1 = (const float*)d_in[23];
  const float* bh1 = (const float*)d_in[24];
  const float* Wh2 = (const float*)d_in[25];
  const float* bh2 = (const float*)d_in[26];

  // --- carve workspace ---
  char* ws = (char*)d_ws;
  auto carve = [&](size_t bytes) {
    void* p = (void*)ws;
    ws += (bytes + 255) & ~(size_t)255;
    return p;
  };
  __bf16* Xb = (__bf16*)carve((size_t)NTOT * 192 * 2);   // node features, bf16
  __bf16* B1 = (__bf16*)carve((size_t)NTOT * HID * 2);   // ping
  __bf16* B2 = (__bf16*)carve((size_t)NTOT * HID * 2);   // pong
  float*  HW = (float*)carve((size_t)NTOT * HID * 4);    // pre-aggregation h@W
  __bf16* Wp_e1[3], *Wp_e2[3], *Wp_g[3], *Wp_h1;
  for (int l = 0; l < 3; ++l) Wp_e1[l] = (__bf16*)carve((size_t)192 * HID * 2);
  for (int l = 0; l < 3; ++l) Wp_e2[l] = (__bf16*)carve((size_t)HID * HID * 2);
  for (int l = 0; l < 3; ++l) Wp_g[l]  = (__bf16*)carve((size_t)HID * HID * 2);
  Wp_h1 = (__bf16*)carve((size_t)HID * HID * 2);

  // --- 1. pack weights into WMMA fragment order ---
  auto pack = [&](const float* W, __bf16* Wp, int K) {
    int threads = (K >> 5) * (HID / 16) * 32;
    pack_weight<<<(threads + 255) / 256, 256, 0, stream>>>(W, Wp, K);
  };
  for (int l = 0; l < 3; ++l) { pack(eW1[l], Wp_e1[l], 192); pack(eW2[l], Wp_e2[l], 256); }
  for (int l = 0; l < 3; ++l) pack(gW[l], Wp_g[l], 256);
  pack(Wh1, Wp_h1, 256);

  // --- 2. NCHW -> [node, DIM] bf16 ---
  for (int l = 0; l < 3; ++l) {
    dim3 g(LSZ[l] / 64, 192 / 32);
    transpose_feats<<<g, 256, 0, stream>>>(F[l], Xb, LSZ[l], LOFF[l]);
  }

  // --- 3. per-level encoder MLPs (relu(X@W1+b1)@W2+b2) ---
  for (int l = 0; l < 3; ++l) {
    dim3 g(LSZ[l] / 128, HID / 128);
    gemm_wmma<192, true, true, false><<<g, 256, 0, stream>>>(
        Xb + (size_t)LOFF[l] * 192, Wp_e1[l], eB1[l],
        B1 + (size_t)LOFF[l] * HID, LSZ[l]);
    gemm_wmma<256, true, false, false><<<g, 256, 0, stream>>>(
        B1 + (size_t)LOFF[l] * HID, Wp_e2[l], eB2[l],
        B2 + (size_t)LOFF[l] * HID, LSZ[l]);
  }

  // --- 4. three GCN layers: GEMM (all nodes) then per-level stencil ---
  __bf16* cur = B2;
  __bf16* nxt = B1;
  dim3 gAll(NTOT / 128, HID / 128);
  for (int i = 0; i < 3; ++i) {
    gemm_wmma<256, false, false, true><<<gAll, 256, 0, stream>>>(
        cur, Wp_g[i], nullptr, HW, NTOT);
    for (int l = 0; l < 3; ++l) {
      int threads = LSZ[l] * 64;
      gcn_aggregate<<<(threads + 255) / 256, 256, 0, stream>>>(
          HW, gB[i], nxt, LH[l], LW[l], LOFF[l]);
    }
    __bf16* t = cur; cur = nxt; nxt = t;
  }

  // --- 5. head: relu(h@Wh1+bh1) then dot with Wh2 + bh2 -> d_out ---
  gemm_wmma<256, true, true, false><<<gAll, 256, 0, stream>>>(
      cur, Wp_h1, bh1, nxt, NTOT);
  head_gemv<<<NTOT / 8, 256, 0, stream>>>(nxt, Wh2, bh2, (float*)d_out, NTOT);
}
